// SimpleCrossAttention_75024488727182
// MI455X (gfx1250) — matrix operand order repaired and checked by reading, not compiled, not run on hardware
//
#include <hip/hip_runtime.h>

typedef __attribute__((ext_vector_type(16))) __bf16 v16bf;
typedef __attribute__((ext_vector_type(8)))  float  v8f;

#define NHEAD 16
#define DH 64
#define DMODEL 1024

__device__ __forceinline__ unsigned short f2bf(float f) {
  unsigned u = __float_as_uint(f);
  unsigned r = 0x7FFFu + ((u >> 16) & 1u);   // round-to-nearest-even
  return (unsigned short)((u + r) >> 16);
}

// 16-byte async global->LDS copy (per lane). ASYNCcnt-tracked.
__device__ __forceinline__ void async16(const unsigned short* lds_p, const void* g) {
  unsigned lds = (unsigned)(uintptr_t)lds_p;   // low 32 bits = LDS offset
  asm volatile("global_load_async_to_lds_b128 %0, %1, off"
               :: "v"(lds), "v"(g) : "memory");
}
__device__ __forceinline__ void wait_async_le4() {
  asm volatile("s_wait_asynccnt 0x4" ::: "memory");
}
__device__ __forceinline__ void wait_async_le2() {
  asm volatile("s_wait_asynccnt 0x2" ::: "memory");
}
__device__ __forceinline__ void wait_async_0() {
  asm volatile("s_wait_asynccnt 0x0" ::: "memory");
}

// ---------------------------------------------------------------------------
// Fragment loaders (work for global or LDS pointers; LDS emits ds_load_b128).
// A fragment 16x32 bf16 from row-major [..][ld] at (m0,k0). ISA A-layout:
// lane l (m=l&15,h=l>>4): elems 0..7 -> K=h*8+e; 8..15 -> K=16+h*8+(e-8).
__device__ __forceinline__ v16bf load_frag_a(const unsigned short* base, int ld,
                                             int m0, int k0) {
  int lane = threadIdx.x & 31;
  int m = lane & 15, h = lane >> 4;
  const unsigned short* p = base + (size_t)(m0 + m) * ld + k0 + h * 8;
  v16bf r;
  reinterpret_cast<uint4*>(&r)[0] = *reinterpret_cast<const uint4*>(p);
  reinterpret_cast<uint4*>(&r)[1] = *reinterpret_cast<const uint4*>(p + 16);
  return r;
}
// B fragment 32x16 from B^T row-major [N][ld]: lane l (n=l&15,h=l>>4):
// elems 0..7 -> K=h*8+e; 8..15 -> K=16+h*8+(e-8).
__device__ __forceinline__ v16bf load_frag_bt(const unsigned short* base, int ld,
                                              int n0, int k0) {
  int lane = threadIdx.x & 31;
  int n = lane & 15, h = lane >> 4;
  const unsigned short* p = base + (size_t)(n0 + n) * ld + k0 + h * 8;
  v16bf r;
  reinterpret_cast<uint4*>(&r)[0] = *reinterpret_cast<const uint4*>(p);
  reinterpret_cast<uint4*>(&r)[1] = *reinterpret_cast<const uint4*>(p + 16);
  return r;
}

// ---------------------------------------------------------------------------
// LayerNorm fp32 -> bf16 bits, one block per row.
__global__ void ln_bf16_kernel(const float* __restrict__ x,
                               const float* __restrict__ g,
                               const float* __restrict__ b,
                               unsigned short* __restrict__ y, int cols) {
  int row = blockIdx.x;
  const float* xr = x + (size_t)row * cols;
  unsigned short* yr = y + (size_t)row * cols;
  float s = 0.f, ss = 0.f;
  for (int i = threadIdx.x; i < cols; i += blockDim.x) {
    float v = xr[i]; s += v; ss += v * v;
  }
  __shared__ float red[16];
  #pragma unroll
  for (int off = 16; off > 0; off >>= 1) {
    s  += __shfl_xor(s,  off, 32);
    ss += __shfl_xor(ss, off, 32);
  }
  int wv = threadIdx.x >> 5;
  if ((threadIdx.x & 31) == 0) { red[wv] = s; red[8 + wv] = ss; }
  __syncthreads();
  s = 0.f; ss = 0.f;
  #pragma unroll
  for (int i = 0; i < 8; ++i) { s += red[i]; ss += red[8 + i]; }
  float mu  = s * (1.0f / cols);
  float var = ss * (1.0f / cols) - mu * mu;
  float inv = rsqrtf(var + 1e-5f);
  for (int i = threadIdx.x; i < cols; i += blockDim.x)
    yr[i] = f2bf((xr[i] - mu) * inv * g[i] + b[i]);
}

// ---------------------------------------------------------------------------
// fp32 W[K][N] -> bf16 W^T[N][K] (LDS-tiled transpose, block 32x8)
__global__ void wconvT_kernel(const float* __restrict__ w,
                              unsigned short* __restrict__ wt, int K, int N) {
  __shared__ float tile[32][33];
  int kb = blockIdx.y * 32, nb = blockIdx.x * 32;
  int tx = threadIdx.x, ty = threadIdx.y;
  #pragma unroll
  for (int i = 0; i < 32; i += 8)
    tile[ty + i][tx] = w[(size_t)(kb + ty + i) * N + (nb + tx)];
  __syncthreads();
  #pragma unroll
  for (int i = 0; i < 32; i += 8)
    wt[(size_t)(nb + ty + i) * K + (kb + tx)] = f2bf(tile[tx][ty + i]);
}

// ---------------------------------------------------------------------------
// C = A(bf16 [M][K]) * B (as B^T bf16 [N][K]).
// Block = 8 waves = 128x128 C tile; K staged 32 at a time into LDS with
// async-to-LDS double buffering. Wave w: rows 16w..16w+15, all 128 cols.
// mode 0: bf16 [M][N]; 1: bf16 [b][h][tok][DH]; 2: bf16 [b][h][DH][tok];
// mode 3: fp32 [M][N]
#define GM 128
#define GN 128
#define GK 32
__global__ __launch_bounds__(256) void
gemm_wmma_kernel(const unsigned short* __restrict__ A,
                 const unsigned short* __restrict__ BT,
                 void* __restrict__ out,
                 int M, int N, int K, int mode, int ntok) {
  __shared__ unsigned short As[2][GM][GK];   // 2 x 8KB
  __shared__ unsigned short Bs[2][GN][GK];   // 2 x 8KB
  int tid = threadIdx.x;
  int wv  = tid >> 5;
  int tilesN = N / GN;
  int bm = blockIdx.x / tilesN, bn = blockIdx.x - bm * tilesN;
  int m0 = bm * GM, n0 = bn * GN;
  int nkt = K / GK;

  auto issue = [&](int kt, int buf) {
    int k0 = kt * GK;
    // A tile: 128 rows x 64B = 512 x 16B chunks, 2 per thread
    #pragma unroll
    for (int it = 0; it < 2; ++it) {
      int c = tid + it * 256;
      int row = c >> 2, part = c & 3;
      async16(&As[buf][row][part * 8],
              A + (size_t)(m0 + row) * K + k0 + part * 8);
    }
    // B tile: 128 rows x 64B = 512 x 16B chunks, 2 per thread
    #pragma unroll
    for (int it = 0; it < 2; ++it) {
      int c = tid + it * 256;
      int row = c >> 2, part = c & 3;
      async16(&Bs[buf][row][part * 8],
              BT + (size_t)(n0 + row) * K + k0 + part * 8);
    }
  };

  v8f acc[8] = {};
  issue(0, 0);
  for (int kt = 0; kt < nkt; ++kt) {
    int buf = kt & 1;
    if (kt + 1 < nkt) { issue(kt + 1, buf ^ 1); wait_async_le4(); }
    else              { wait_async_0(); }
    __syncthreads();
    v16bf a = load_frag_a(&As[buf][0][0], GK, wv * 16, 0);
    #pragma unroll
    for (int j = 0; j < 8; ++j) {
      v16bf bm_ = load_frag_bt(&Bs[buf][0][0], GK, j * 16, 0);
      acc[j] = __builtin_amdgcn_wmma_f32_16x16x32_bf16(false, a, false, bm_,
                                                       (short)0, acc[j], false, false);
    }
    __syncthreads();
  }

  int lane = tid & 31;
  int half = lane >> 4, nn = lane & 15;
  int mbase = m0 + wv * 16;
  #pragma unroll
  for (int j = 0; j < 8; ++j) {
    #pragma unroll
    for (int r = 0; r < 8; ++r) {
      int m = mbase + r + 8 * half;
      int n = n0 + j * 16 + nn;
      float v = acc[j][r];
      if (mode == 3) {
        ((float*)out)[(size_t)m * N + n] = v;
      } else {
        size_t idx;
        if (mode == 0) {
          idx = (size_t)m * N + n;
        } else {
          int bb = m / ntok, tok = m - bb * ntok;
          int hh = n >> 6, d = n & 63;
          if (mode == 1) idx = ((size_t)(bb * NHEAD + hh) * ntok + tok) * DH + d;
          else           idx = ((size_t)(bb * NHEAD + hh) * DH + d) * ntok + tok;
        }
        ((unsigned short*)out)[idx] = f2bf(v);
      }
    }
  }
}

// ---------------------------------------------------------------------------
// Flash attention. q,k: bf16 [b*h][tok][DH]; vt: bf16 [b*h][DH][tok];
// z: bf16 [b][tok][DMODEL]. Block = 8 waves = 128 q rows of one (b,h);
// 32-token K/V chunks async-staged into LDS, double-buffered, shared by waves.
__global__ __launch_bounds__(256) void
attn_wmma_kernel(const unsigned short* __restrict__ q,
                 const unsigned short* __restrict__ k,
                 const unsigned short* __restrict__ vt,
                 unsigned short* __restrict__ z,
                 int nq, int nk) {
  __shared__ unsigned short Ks[2][32][DH];   // [buf][token][d]  2 x 4KB
  __shared__ unsigned short Vs[2][DH][32];   // [buf][d][token]  2 x 4KB
  __shared__ unsigned short Ps[8][16 * 32];  // per-wave P tile  8KB
  int tid = threadIdx.x;
  int lane = tid & 31;
  int wv = tid >> 5;
  int qblocks = nq / 128;
  int bh = blockIdx.x / qblocks;
  int qb = blockIdx.x - bh * qblocks;
  const unsigned short* qp = q  + (size_t)bh * nq * DH;
  const unsigned short* kp = k  + (size_t)bh * nk * DH;
  const unsigned short* vp = vt + (size_t)bh * DH * nk;
  int m0 = qb * 128 + wv * 16;
  int half = lane >> 4, nn = lane & 15;

  auto issue = [&](int kc, int buf) {
    { // K chunk: 32 tokens x 128B = 256 x 16B chunks, 1 per thread
      int row = tid >> 3, part = tid & 7;
      async16(&Ks[buf][row][part * 8],
              kp + (size_t)(kc + row) * DH + part * 8);
    }
    { // V chunk: 64 d-rows x 64B = 256 x 16B chunks, 1 per thread
      int d = tid >> 2, part = tid & 3;
      async16(&Vs[buf][d][part * 8],
              vp + (size_t)d * nk + kc + part * 8);
    }
  };

  v16bf qa0 = load_frag_a(qp, DH, m0, 0);
  v16bf qa1 = load_frag_a(qp, DH, m0, 32);

  v8f o[4] = {};
  float mrow[8], lrow[8];
  #pragma unroll
  for (int r = 0; r < 8; ++r) { mrow[r] = -3.0e38f; lrow[r] = 0.f; }

  const float SCL = 0.125f * 1.44269504088896340736f;  // DH^-0.5 * log2(e)
  unsigned short* pb = &Ps[wv][0];
  int nchunks = nk / 32;

  issue(0, 0);
  for (int ct = 0; ct < nchunks; ++ct) {
    int buf = ct & 1;
    if (ct + 1 < nchunks) { issue((ct + 1) * 32, buf ^ 1); wait_async_le2(); }
    else                  { wait_async_0(); }
    __syncthreads();

    const unsigned short* kb = &Ks[buf][0][0];
    const unsigned short* vb = &Vs[buf][0][0];
    v8f s0 = {}, s1 = {};
    s0 = __builtin_amdgcn_wmma_f32_16x16x32_bf16(false, qa0, false,
            load_frag_bt(kb, DH, 0, 0),   (short)0, s0, false, false);
    s0 = __builtin_amdgcn_wmma_f32_16x16x32_bf16(false, qa1, false,
            load_frag_bt(kb, DH, 0, 32),  (short)0, s0, false, false);
    s1 = __builtin_amdgcn_wmma_f32_16x16x32_bf16(false, qa0, false,
            load_frag_bt(kb, DH, 16, 0),  (short)0, s1, false, false);
    s1 = __builtin_amdgcn_wmma_f32_16x16x32_bf16(false, qa1, false,
            load_frag_bt(kb, DH, 16, 32), (short)0, s1, false, false);

    // online softmax; rows of a fragment live across 16-lane halves
    #pragma unroll
    for (int r = 0; r < 8; ++r) {
      float t0 = s0[r] * SCL;
      float t1 = s1[r] * SCL;
      float tmax = fmaxf(t0, t1);
      #pragma unroll
      for (int off = 8; off > 0; off >>= 1)
        tmax = fmaxf(tmax, __shfl_xor(tmax, off, 16));
      float nm   = fmaxf(mrow[r], tmax);
      float corr = exp2f(mrow[r] - nm);
      float p0   = exp2f(t0 - nm);
      float p1   = exp2f(t1 - nm);
      float rs = p0 + p1;
      #pragma unroll
      for (int off = 8; off > 0; off >>= 1)
        rs += __shfl_xor(rs, off, 16);
      lrow[r] = lrow[r] * corr + rs;
      mrow[r] = nm;
      o[0][r] *= corr; o[1][r] *= corr; o[2][r] *= corr; o[3][r] *= corr;
      int m = r + 8 * half;
      pb[m * 32 + nn]      = f2bf(p0);
      pb[m * 32 + 16 + nn] = f2bf(p1);
    }

    // C-layout -> A-layout via per-wave LDS (same-wave DS ops are in-order)
    int am = lane & 15;
    v16bf pa;
    reinterpret_cast<uint4*>(&pa)[0] =
        *reinterpret_cast<const uint4*>(&pb[am * 32 + half * 8]);
    reinterpret_cast<uint4*>(&pa)[1] =
        *reinterpret_cast<const uint4*>(&pb[am * 32 + 16 + half * 8]);

    #pragma unroll
    for (int j = 0; j < 4; ++j)
      o[j] = __builtin_amdgcn_wmma_f32_16x16x32_bf16(false, pa, false,
              load_frag_bt(vb, 32, j * 16, 0), (short)0, o[j], false, false);

    __syncthreads();
  }

  int b = bh >> 4, hh = bh & 15;
  #pragma unroll
  for (int j = 0; j < 4; ++j) {
    #pragma unroll
    for (int r = 0; r < 8; ++r) {
      int tok = m0 + r + 8 * half;
      int d = j * 16 + nn;
      float v = o[j][r] / lrow[r];
      z[((size_t)b * nq + tok) * DMODEL + hh * DH + d] = f2bf(v);
    }
  }
}

// ---------------------------------------------------------------------------
extern "C" void kernel_launch(void* const* d_in, const int* in_sizes, int n_in,
                              void* d_out, int out_size, void* d_ws, size_t ws_size,
                              hipStream_t stream) {
  const float* Q    = (const float*)d_in[0];
  const float* KV   = (const float*)d_in[1];
  const float* Wq   = (const float*)d_in[2];
  const float* Wk   = (const float*)d_in[3];
  const float* Wv   = (const float*)d_in[4];
  const float* Wo   = (const float*)d_in[5];
  const float* g_q  = (const float*)d_in[6];
  const float* b_q  = (const float*)d_in[7];
  const float* g_kv = (const float*)d_in[8];
  const float* b_kv = (const float*)d_in[9];

  const int B = 2, NQ = 2048, NK = 2048, D = DMODEL;
  const int MQ = B * NQ;   // 4096 query rows
  const int MK = B * NK;   // 4096 kv rows

  unsigned short* ws = (unsigned short*)d_ws;
  size_t off = 0;
  unsigned short* qn  = ws + off; off += (size_t)MQ * D;
  unsigned short* kvn = ws + off; off += (size_t)MK * D;
  unsigned short* qh  = ws + off; off += (size_t)MQ * D;  // [b][h][tok][DH]
  unsigned short* kh  = ws + off; off += (size_t)MK * D;  // [b][h][tok][DH]
  unsigned short* vth = ws + off; off += (size_t)MK * D;  // [b][h][DH][tok]
  unsigned short* zb  = ws + off; off += (size_t)MQ * D;  // [b][tok][D]
  unsigned short* WqT = ws + off; off += (size_t)D * D;
  unsigned short* WkT = ws + off; off += (size_t)D * D;
  unsigned short* WvT = ws + off; off += (size_t)D * D;
  unsigned short* WoT = ws + off; off += (size_t)D * D;

  dim3 tb(32, 8), tg(D / 32, D / 32);
  wconvT_kernel<<<tg, tb, 0, stream>>>(Wq, WqT, D, D);
  wconvT_kernel<<<tg, tb, 0, stream>>>(Wk, WkT, D, D);
  wconvT_kernel<<<tg, tb, 0, stream>>>(Wv, WvT, D, D);
  wconvT_kernel<<<tg, tb, 0, stream>>>(Wo, WoT, D, D);

  ln_bf16_kernel<<<MQ, 256, 0, stream>>>(Q,  g_q,  b_q,  qn,  D);
  ln_bf16_kernel<<<MK, 256, 0, stream>>>(KV, g_kv, b_kv, kvn, D);

  int gemm_blocks = (MQ / GM) * (D / GN);    // 32 * 8 = 256 blocks
  gemm_wmma_kernel<<<gemm_blocks, 256, 0, stream>>>(qn,  WqT, qh,  MQ, D, D, 1, NQ);
  gemm_wmma_kernel<<<gemm_blocks, 256, 0, stream>>>(kvn, WkT, kh,  MK, D, D, 1, NK);
  gemm_wmma_kernel<<<gemm_blocks, 256, 0, stream>>>(kvn, WvT, vth, MK, D, D, 2, NK);

  int attn_blocks = B * NHEAD * (NQ / 128);  // 2*16*16 = 512 blocks
  attn_wmma_kernel<<<attn_blocks, 256, 0, stream>>>(qh, kh, vth, zb, NQ, NK);

  gemm_wmma_kernel<<<gemm_blocks, 256, 0, stream>>>(zb, WoT, d_out, MQ, D, D, 3, NQ);
}
